// SelfAttention_69114613728190
// MI455X (gfx1250) — matrix operand order repaired and checked by reading, not compiled, not run on hardware
//
#include <hip/hip_runtime.h>

#define EMBED 1024
#define HEADS 16
#define HD 64
#define NB 4
#define SEQ 2048

typedef __attribute__((ext_vector_type(16))) __bf16 v16bf;
typedef __attribute__((ext_vector_type(8)))  float  v8f;

#if __has_builtin(__builtin_amdgcn_global_load_async_to_lds_b128) && \
    __has_builtin(__builtin_amdgcn_s_wait_asynccnt)
#define HAVE_ASYNC_LDS 1
#else
#define HAVE_ASYNC_LDS 0
#endif

// Exact parameter types of the async-LDS builtin: pointers to 16-byte int
// vectors in the global (AS1) and LDS (AS3) address spaces.
typedef int v4i_raw __attribute__((vector_size(16)));
typedef __attribute__((address_space(1))) v4i_raw* g_v4i_ptr;
typedef __attribute__((address_space(3))) v4i_raw* l_v4i_ptr;

// 16-byte global -> LDS copy. Async (ASYNCcnt-tracked, no VGPR round trip) when
// the gfx1250 builtin exists, otherwise a plain sync copy with identical
// barrier semantics.
__device__ inline void copy16_g2l(const void* g, void* l) {
#if HAVE_ASYNC_LDS
  __builtin_amdgcn_global_load_async_to_lds_b128((g_v4i_ptr)g, (l_v4i_ptr)l, 0, 0);
#else
  *(uint4*)l = *(const uint4*)g;
#endif
}

#if HAVE_ASYNC_LDS
#define WAIT_ASYNC(n) __builtin_amdgcn_s_wait_asynccnt(n)
#else
#define WAIT_ASYNC(n)
#endif

// ---------------------------------------------------------------------------
// WMMA helper: D = A(16x32 bf16) x B(32x16 bf16) + C(16x16 f32)
// ---------------------------------------------------------------------------
__device__ inline v8f wmma_bf16(v16bf a, v16bf b, v8f c) {
  return __builtin_amdgcn_wmma_f32_16x16x32_bf16(
      /*neg_a=*/false, a, /*neg_b=*/false, b,
      /*c_mod=*/(short)0, c, /*reuse_a=*/false, /*reuse_b=*/false);
}

// A-fragment (16x32, bf16), ISA 7.12.2 layout. Per lane the 16 slots are two
// contiguous 8-element runs -> compiler merges into 2x ds_load_b128.
__device__ inline v16bf load_a_frag(const __bf16* __restrict__ tile, int ld, int lane) {
  int m = lane & 15, half = lane >> 4;
  v16bf a;
#pragma unroll
  for (int i = 0; i < 16; ++i) {
    int j = i >> 1, p = i & 1;
    int k = ((j & 4) ? 16 : 0) + ((j & 3) * 2 + p) + half * 8;
    a[i] = tile[m * ld + k];
  }
  return a;
}

// B-fragment where B[k][n] = tile[n][k]  (tile row-major NxK, ld = row stride).
// Slot i = tile[n*ld + i + 8*half + (i>=8 ? 8 : 0)]: two 8-runs -> 2x ds_load_b128 per frag.
// ALL B operands below are stored in LDS in this "n-major" layout.
__device__ inline v16bf load_b_frag_trans(const __bf16* __restrict__ tile, int ld, int lane) {
  int n = lane & 15, half = lane >> 4;
  v16bf b;
#pragma unroll
  for (int i = 0; i < 16; ++i)
    b[i] = tile[n * ld + i + half * 8 + ((i >> 3) << 3)];
  return b;
}

// ---------------------------------------------------------------------------
// Kernel 1: per-head projections  out = x @ W.T   (bf16 out)
//   q,k: [N,H,S,64] (row-major per head)     v: [N,H,64,S] (d-major, so the
//   attention kernel's P·V B-fragments become contiguous gathers)
// grid = (SEQ/128, 3*NB*HEADS), block = 256 (8 waves; each wave 16 rows x 64)
// ---------------------------------------------------------------------------
__global__ __launch_bounds__(256) void proj_kernel(
    const float* __restrict__ q_in, const float* __restrict__ k_in,
    const float* __restrict__ v_in,
    const float* __restrict__ Wq, const float* __restrict__ Wk,
    const float* __restrict__ Wv,
    __bf16* __restrict__ qw, __bf16* __restrict__ kw, __bf16* __restrict__ vw)
{
  __shared__ alignas(16) __bf16 Wl[HD * HD];
  __shared__ alignas(16) __bf16 Xl[128 * HD];

  int tid  = threadIdx.x;
  int s0   = blockIdx.x * 128;
  int mnh  = blockIdx.y;
  int mat  = mnh / (NB * HEADS);
  int nh   = mnh % (NB * HEADS);
  int n    = nh / HEADS, h = nh % HEADS;

  const float* X = (mat == 0) ? q_in : (mat == 1) ? k_in : v_in;
  const float* W = (mat == 0) ? Wq   : (mat == 1) ? Wk   : Wv;

  // W (64x64 f32, contiguous) -> bf16 LDS
  for (int i = tid; i < HD * HD / 4; i += 256) {
    float4 f = ((const float4*)W)[i];
    Wl[4 * i + 0] = (__bf16)f.x; Wl[4 * i + 1] = (__bf16)f.y;
    Wl[4 * i + 2] = (__bf16)f.z; Wl[4 * i + 3] = (__bf16)f.w;
  }
  // X tile: 128 rows x 64 head-cols (each row contiguous 64 f32)
  for (int i = tid; i < 128 * HD / 4; i += 256) {
    int row = i >> 4, c4 = i & 15;
    float4 f = *(const float4*)(X + (size_t)(n * SEQ + s0 + row) * EMBED + h * HD + c4 * 4);
    __bf16* dst = &Xl[row * HD + c4 * 4];
    dst[0] = (__bf16)f.x; dst[1] = (__bf16)f.y; dst[2] = (__bf16)f.z; dst[3] = (__bf16)f.w;
  }
  __syncthreads();

  int wave = tid >> 5, lane = tid & 31;
  int nn = lane & 15, half = lane >> 4;
  v16bf a0 = load_a_frag(&Xl[wave * 16 * HD],      HD, lane);
  v16bf a1 = load_a_frag(&Xl[wave * 16 * HD + 32], HD, lane);

#pragma unroll
  for (int nt = 0; nt < 4; ++nt) {
    // B[k][e] = W[e][k]  (x @ W.T)
    v16bf b0 = load_b_frag_trans(&Wl[(nt * 16) * HD],      HD, lane);
    v16bf b1 = load_b_frag_trans(&Wl[(nt * 16) * HD + 32], HD, lane);
    v8f acc = {};
    acc = wmma_bf16(a0, b0, acc);
    acc = wmma_bf16(a1, b1, acc);
    if (mat == 2) {  // V: d-major layout [nh][d][s]
#pragma unroll
      for (int r = 0; r < 8; ++r) {
        int row = s0 + wave * 16 + r + half * 8;
        vw[((size_t)nh * HD + nt * 16 + nn) * SEQ + row] = (__bf16)acc[r];
      }
    } else {
      __bf16* O = (mat == 0) ? qw : kw;
#pragma unroll
      for (int r = 0; r < 8; ++r) {
        int row = s0 + wave * 16 + r + half * 8;
        O[((size_t)nh * SEQ + row) * HD + nt * 16 + nn] = (__bf16)acc[r];
      }
    }
  }
}

// ---------------------------------------------------------------------------
// Kernel 2: flash attention over one (n,h); block = 128 q rows, 8 waves.
// K/V tiles double-buffered via global_load_async_to_lds_b128 (ASYNCcnt).
// grid = (NB*HEADS, SEQ/128), block = 256
// ---------------------------------------------------------------------------
__global__ __launch_bounds__(256) void fattn_kernel(
    const __bf16* __restrict__ qw, const __bf16* __restrict__ kw,
    const __bf16* __restrict__ vw, const int* __restrict__ mask,
    __bf16* __restrict__ ow)
{
  __shared__ alignas(16) __bf16 Ql[128 * HD];        // 16 KB
  __shared__ alignas(16) __bf16 Kl[2][32 * HD];      // 2 x 4 KB, [k][d]
  __shared__ alignas(16) __bf16 Vl[2][HD * 32];      // 2 x 4 KB, [d][k]
  __shared__ alignas(16) __bf16 Pl[8 * 16 * 32];     //  8 KB (per-wave P staging)

  int tid = threadIdx.x;
  int nh  = blockIdx.x;
  int n   = nh / HEADS, h = nh % HEADS;
  int q0  = blockIdx.y * 128;

  const __bf16* Qp  = qw + ((size_t)nh * SEQ + q0) * HD;
  const __bf16* Kp  = kw + (size_t)nh * SEQ * HD;       // [s][d]
  const __bf16* Vth = vw + (size_t)nh * HD * SEQ;       // [d][s]

  // per-thread fill chunks (16 B each; 256 threads cover a full 4 KB tile)
  int vd = tid >> 2, vp = tid & 3;  // V: row d = tid/4, 8-elem part within 32 k

  // prologue: kick off buffer 0 (k0 = 0)
  copy16_g2l(Kp + (size_t)0 * HD + tid * 8, &Kl[0][tid * 8]);
  copy16_g2l(Vth + (size_t)vd * SEQ + 0 + vp * 8, &Vl[0][vd * 32 + vp * 8]);

  // Q tile is contiguous (16 KB) -> 128-bit copies
  {
    const uint4* src = (const uint4*)Qp;
    uint4* dst = (uint4*)Ql;
    for (int i = tid; i < 128 * HD * 2 / 16; i += 256) dst[i] = src[i];
  }
  __syncthreads();

  int wave = tid >> 5, lane = tid & 31;
  int nn = lane & 15, half = lane >> 4;
  v16bf aq0 = load_a_frag(&Ql[wave * 16 * HD],      HD, lane);
  v16bf aq1 = load_a_frag(&Ql[wave * 16 * HD + 32], HD, lane);

  v8f acc0 = {}, acc1 = {}, acc2 = {}, acc3 = {};
  float mrow[8], lrow[8];
#pragma unroll
  for (int r = 0; r < 8; ++r) { mrow[r] = -1e30f; lrow[r] = 0.0f; }

  const float scale = 0.03125f;  // 1/sqrt(EMBED) = 1/32  (matches reference)
  const int NIT = SEQ / 32;

  for (int it = 0; it < NIT; ++it) {
    int cur = it & 1;
    int k0  = it * 32;
    if (it + 1 < NIT) {
      // fill the other buffer (all waves finished reading it at the barrier
      // that ended iteration it-1)
      int nxt = 1 - cur, kn = k0 + 32;
      copy16_g2l(Kp + (size_t)kn * HD + tid * 8, &Kl[nxt][tid * 8]);
      copy16_g2l(Vth + (size_t)vd * SEQ + kn + vp * 8, &Vl[nxt][vd * 32 + vp * 8]);
      WAIT_ASYNC(2);   // the 2 newest async ops are the fill just issued;
                       // everything older (current buffer) has landed
    } else {
      WAIT_ASYNC(0);
    }
    __syncthreads();   // all waves' fills for buf[cur] complete

    const __bf16* Kb = Kl[cur];
    const __bf16* Vb = Vl[cur];

    // S = Q(16x64) · Kᵀ  -> two 16x16 f32 tiles (keys k0..+15, k0+16..+31)
    v8f s0 = {}, s1 = {};
    s0 = wmma_bf16(aq0, load_b_frag_trans(&Kb[0],            HD, lane), s0);
    s0 = wmma_bf16(aq1, load_b_frag_trans(&Kb[32],           HD, lane), s0);
    s1 = wmma_bf16(aq0, load_b_frag_trans(&Kb[16 * HD],      HD, lane), s1);
    s1 = wmma_bf16(aq1, load_b_frag_trans(&Kb[16 * HD + 32], HD, lane), s1);

    int msk0 = mask[n * SEQ + k0 + nn];
    int msk1 = mask[n * SEQ + k0 + 16 + nn];

    __bf16* pw = &Pl[wave * 16 * 32];
#pragma unroll
    for (int r = 0; r < 8; ++r) {
      float e0 = (msk0 == 0) ? -1e20f : s0[r] * scale;
      float e1 = (msk1 == 0) ? -1e20f : s1[r] * scale;
      // online softmax; row r+8*half lives across the 16 lanes of this half
      float mx = fmaxf(e0, e1);
#pragma unroll
      for (int d = 1; d < 16; d <<= 1) mx = fmaxf(mx, __shfl_xor(mx, d, 32));
      float mnew = fmaxf(mrow[r], mx);
      float c  = __expf(mrow[r] - mnew);
      float p0 = __expf(e0 - mnew);
      float p1 = __expf(e1 - mnew);
      float rs = p0 + p1;
#pragma unroll
      for (int d = 1; d < 16; d <<= 1) rs += __shfl_xor(rs, d, 32);
      lrow[r] = lrow[r] * c + rs;
      mrow[r] = mnew;
      acc0[r] *= c; acc1[r] *= c; acc2[r] *= c; acc3[r] *= c;
      // stage P (C-layout -> row-major) for A-fragment regather
      pw[(r + half * 8) * 32 + nn]      = (__bf16)p0;
      pw[(r + half * 8) * 32 + 16 + nn] = (__bf16)p1;
    }
    // O += P(16x32) · V(32x64); V is d-major so B-frags are contiguous
    v16bf ap = load_a_frag(pw, 32, lane);
    acc0 = wmma_bf16(ap, load_b_frag_trans(&Vb[ 0 * 32], 32, lane), acc0);
    acc1 = wmma_bf16(ap, load_b_frag_trans(&Vb[16 * 32], 32, lane), acc1);
    acc2 = wmma_bf16(ap, load_b_frag_trans(&Vb[32 * 32], 32, lane), acc2);
    acc3 = wmma_bf16(ap, load_b_frag_trans(&Vb[48 * 32], 32, lane), acc3);

    __syncthreads();   // buf[cur] free for the fill issued at iteration it+1
  }

  // normalize + write O as bf16 into [N,S,EMBED]
#pragma unroll
  for (int r = 0; r < 8; ++r) {
    float inv = 1.0f / lrow[r];
    int row = q0 + wave * 16 + r + half * 8;
    __bf16* dst = ow + ((size_t)n * SEQ + row) * EMBED + h * HD;
    dst[0 * 16 + nn] = (__bf16)(acc0[r] * inv);
    dst[1 * 16 + nn] = (__bf16)(acc1[r] * inv);
    dst[2 * 16 + nn] = (__bf16)(acc2[r] * inv);
    dst[3 * 16 + nn] = (__bf16)(acc3[r] * inv);
  }
}

// ---------------------------------------------------------------------------
// Kernel 3: Y = O @ Wo.T + bo   (8192 x 1024 x 1024), f32 output
// grid = (NB*SEQ/128, EMBED/64), block = 256
// ---------------------------------------------------------------------------
__global__ __launch_bounds__(256) void outproj_kernel(
    const __bf16* __restrict__ ow, const float* __restrict__ Wo,
    const float* __restrict__ bo, float* __restrict__ out)
{
  __shared__ alignas(16) __bf16 Al[128 * 32];  // 8 KB
  __shared__ alignas(16) __bf16 Bl[64 * 32];   // 4 KB, e-major: Bl[e][k]

  int tid  = threadIdx.x;
  int row0 = blockIdx.x * 128;
  int col0 = blockIdx.y * 64;
  int wave = tid >> 5, lane = tid & 31;
  int nn = lane & 15, half = lane >> 4;

  v8f acc0 = {}, acc1 = {}, acc2 = {}, acc3 = {};

  for (int kb = 0; kb < EMBED; kb += 32) {
    __syncthreads();
    // A: 128 rows x 32 k-cols of O (bf16, 64 B per row slice)
    for (int i = tid; i < 128 * 4; i += 256) {
      int row = i >> 2, c = i & 3;
      ((uint4*)Al)[row * 4 + c] =
          *(const uint4*)(ow + (size_t)(row0 + row) * EMBED + kb + c * 8);
    }
    // Bl[e][k] = Wo[col0+e][kb+k]  (f32 -> bf16, contiguous store AND gather)
    for (int i = tid; i < 64 * 8; i += 256) {
      int e = i >> 3, c4 = i & 7;
      float4 f = *(const float4*)(Wo + (size_t)(col0 + e) * EMBED + kb + c4 * 4);
      __bf16* dst = &Bl[e * 32 + c4 * 4];
      dst[0] = (__bf16)f.x; dst[1] = (__bf16)f.y;
      dst[2] = (__bf16)f.z; dst[3] = (__bf16)f.w;
    }
    __syncthreads();

    v16bf a = load_a_frag(&Al[wave * 16 * 32], 32, lane);
    acc0 = wmma_bf16(a, load_b_frag_trans(&Bl[ 0 * 32], 32, lane), acc0);
    acc1 = wmma_bf16(a, load_b_frag_trans(&Bl[16 * 32], 32, lane), acc1);
    acc2 = wmma_bf16(a, load_b_frag_trans(&Bl[32 * 32], 32, lane), acc2);
    acc3 = wmma_bf16(a, load_b_frag_trans(&Bl[48 * 32], 32, lane), acc3);
  }

#pragma unroll
  for (int r = 0; r < 8; ++r) {
    int row = row0 + wave * 16 + r + half * 8;
    float* dst = out + (size_t)row * EMBED + col0;
    dst[0 * 16 + nn] = acc0[r] + bo[col0 + 0 * 16 + nn];
    dst[1 * 16 + nn] = acc1[r] + bo[col0 + 1 * 16 + nn];
    dst[2 * 16 + nn] = acc2[r] + bo[col0 + 2 * 16 + nn];
    dst[3 * 16 + nn] = acc3[r] + bo[col0 + 3 * 16 + nn];
  }
}

// ---------------------------------------------------------------------------
extern "C" void kernel_launch(void* const* d_in, const int* in_sizes, int n_in,
                              void* d_out, int out_size, void* d_ws, size_t ws_size,
                              hipStream_t stream) {
  (void)in_sizes; (void)n_in; (void)out_size; (void)ws_size;
  const float* values = (const float*)d_in[0];
  const float* keys_  = (const float*)d_in[1];
  const float* query  = (const float*)d_in[2];
  const int*   mask   = (const int*)d_in[3];
  const float* Wv = (const float*)d_in[4];
  const float* Wk = (const float*)d_in[5];
  const float* Wq = (const float*)d_in[6];
  const float* Wo = (const float*)d_in[7];
  const float* bo = (const float*)d_in[8];
  float* out = (float*)d_out;

  // ws layout: qws | kws ([N,H,S,64]) | vws ([N,H,64,S]) | ows ([N,S,E]) bf16
  size_t per = (size_t)NB * HEADS * SEQ * HD;  // 8Mi elements
  __bf16* qws = (__bf16*)d_ws;
  __bf16* kws = qws + per;
  __bf16* vws = kws + per;
  __bf16* ows = vws + per;  // NB*SEQ*EMBED == per

  proj_kernel<<<dim3(SEQ / 128, 3 * NB * HEADS), 256, 0, stream>>>(
      query, keys_, values, Wq, Wk, Wv, qws, kws, vws);
  fattn_kernel<<<dim3(NB * HEADS, SEQ / 128), 256, 0, stream>>>(
      qws, kws, vws, mask, ows);
  outproj_kernel<<<dim3(NB * SEQ / 128, EMBED / 64), 256, 0, stream>>>(
      ows, Wo, bo, out);
}
